// BiLSTMDecoderModel_75058848464972
// MI455X (gfx1250) — matrix operand order repaired and checked by reading, not compiled, not run on hardware
//
#include <hip/hip_runtime.h>
#include <math.h>

// ---- problem constants (match reference) ----
#define B_ 128
#define S_ 256
#define E_ 256
#define H_ 512
#define L_ 5
#define FOURH_ (4 * H_)

typedef _Float16 half_t;
typedef __attribute__((ext_vector_type(16))) _Float16 v16h;
typedef __attribute__((ext_vector_type(8)))  _Float16 v8h;
typedef __attribute__((ext_vector_type(8)))  float    v8f;

__device__ __forceinline__ float sigmoidf_(float x) { return 1.0f / (1.0f + __expf(-x)); }

// ---------------------------------------------------------------------------
// WMMA fragment loaders (layouts per CDNA5 ISA 7.12.2, wave32)
// A: 16x32 (MxK) f16, row-major source with leading dim ld.
//   lane L (row = L&15): VGPR0-3 hold K = kb..kb+7, VGPR4-7 hold K = 16+kb..16+kb+7,
//   kb = (L<16) ? 0 : 8  -> two contiguous 16B chunks.
// B: 32x16 (KxN) f16 = W^T of row-major W(N,K):
//   lane L (col = L&15): same K layout as A (K = kb..kb+7, 16+kb..16+kb+7)
//   -> two contiguous 16B chunks from W row (n0+col).
// ---------------------------------------------------------------------------
__device__ __forceinline__ v16h load_a_frag(const half_t* __restrict__ x, int ld,
                                            int m0, int k0, int row, int hi) {
    const half_t* p = x + (size_t)(m0 + row) * ld + k0 + hi * 8;
    v8h lo = *(const v8h*)(p);
    v8h hh = *(const v8h*)(p + 16);
    v16h a;
#pragma unroll
    for (int i = 0; i < 8; ++i) { a[i] = lo[i]; a[i + 8] = hh[i]; }
    return a;
}

__device__ __forceinline__ v16h load_b_frag(const half_t* __restrict__ w, int ld,
                                            int n0, int k0, int col, int hi) {
    const half_t* p = w + (size_t)(n0 + col) * ld + k0 + hi * 8;
    v8h lo = *(const v8h*)(p);
    v8h hh = *(const v8h*)(p + 16);
    v16h b;
#pragma unroll
    for (int i = 0; i < 8; ++i) { b[i] = lo[i]; b[i + 8] = hh[i]; }
    return b;
}

// ---------------------------------------------------------------------------
// Per-step fused GEMM for BOTH directions:
//   g[dir] = [x_t | h] @ [W_ih | W_hh]^T + (b_ih + b_hh)      (128 x 2048, K=768)
// Each wave computes a 16x32 output tile (two 16x16 N-tiles) so the A fragment
// is reused across two WMMAs (halves A traffic, two independent WMMA chains).
// 1024 wave-tasks: dir(2) x mtile(8) x npair(64); 8 waves/block -> 128 blocks.
// 48 V_WMMA_F32_16X16X32_F16 per wave.
// ---------------------------------------------------------------------------
__global__ __launch_bounds__(256) void lstm_step_gemm(
    const half_t* __restrict__ seq_emb,  // (S,B,E) f16
    const half_t* __restrict__ wih16,    // 2 x (4H,E) f16
    const half_t* __restrict__ whh16,    // 2 x (4H,H) f16
    const half_t* __restrict__ h16,      // 2 x (B,H) f16
    const float*  __restrict__ bias,     // 2 x 4H (b_ih + b_hh)
    float* __restrict__ g,               // 2 x (B,4H) f32
    int t)
{
    const int wave = threadIdx.x >> 5;
    const int lane = threadIdx.x & 31;
    const int row  = lane & 15;
    const int hi   = lane >> 4;

    const int task = blockIdx.x * 8 + wave;      // 0..1023
    const int dir  = task >> 9;
    const int rem  = task & 511;
    const int m0   = (rem >> 6) << 4;            // batch tile (8)
    const int n0   = (rem & 63) << 5;            // 32-wide gate-feature slab (64)

    const int sidx = dir ? ((S_ - t) & (S_ - 1)) : t;   // backward uses -t mod S
    const half_t* x   = seq_emb + (size_t)sidx * B_ * E_;
    const half_t* Wih = wih16 + (size_t)dir * FOURH_ * E_;
    const half_t* Whh = whh16 + (size_t)dir * FOURH_ * H_;
    const half_t* h   = h16   + (size_t)dir * B_ * H_;

    // warm the recurrent-weight rows (L2-resident after step 0)
    __builtin_prefetch(Whh + (size_t)(n0 + row) * H_, 0, 1);
    __builtin_prefetch(Whh + (size_t)(n0 + 16 + row) * H_, 0, 1);

    v8f acc0 = {}, acc1 = {};
#pragma unroll
    for (int k0 = 0; k0 < E_; k0 += 32) {        // input-projection part (K=256)
        v16h a  = load_a_frag(x, E_, m0, k0, row, hi);
        v16h b0 = load_b_frag(Wih, E_, n0,      k0, row, hi);
        v16h b1 = load_b_frag(Wih, E_, n0 + 16, k0, row, hi);
        acc0 = __builtin_amdgcn_wmma_f32_16x16x32_f16(false, a, false, b0,
                                                      (short)0, acc0, false, false);
        acc1 = __builtin_amdgcn_wmma_f32_16x16x32_f16(false, a, false, b1,
                                                      (short)0, acc1, false, false);
    }
#pragma unroll 4
    for (int k0 = 0; k0 < H_; k0 += 32) {        // recurrent part (K=512)
        v16h a  = load_a_frag(h, H_, m0, k0, row, hi);
        v16h b0 = load_b_frag(Whh, H_, n0,      k0, row, hi);
        v16h b1 = load_b_frag(Whh, H_, n0 + 16, k0, row, hi);
        acc0 = __builtin_amdgcn_wmma_f32_16x16x32_f16(false, a, false, b0,
                                                      (short)0, acc0, false, false);
        acc1 = __builtin_amdgcn_wmma_f32_16x16x32_f16(false, a, false, b1,
                                                      (short)0, acc1, false, false);
    }

    const float bn0 = bias[dir * FOURH_ + n0 + row];
    const float bn1 = bias[dir * FOURH_ + n0 + 16 + row];
    float* gout = g + (size_t)dir * B_ * FOURH_;
#pragma unroll
    for (int v = 0; v < 8; ++v) {
        const size_t r = (size_t)(m0 + hi * 8 + v) * FOURH_;
        gout[r + n0 + row]      = acc0[v] + bn0;
        gout[r + n0 + 16 + row] = acc1[v] + bn1;
    }
}

// ---------------------------------------------------------------------------
// LSTM gate nonlinearity + state update, both directions (2*B*H threads).
// Writes next-step h as f16 and the hidden-state history (hs_f / reversed hs_b).
// ---------------------------------------------------------------------------
__global__ __launch_bounds__(256) void lstm_gates(
    const float* __restrict__ g, float* __restrict__ c, half_t* __restrict__ h16,
    float* __restrict__ hsf, float* __restrict__ hsb, int t)
{
    const int i = blockIdx.x * 256 + threadIdx.x;   // < 2*B*H
    const int dir = i >= B_ * H_;
    const int r = i - dir * B_ * H_;
    const int b = r >> 9;           // H = 512
    const int h = r & (H_ - 1);

    const float* gd = g + (size_t)dir * B_ * FOURH_ + (size_t)b * FOURH_;
    const float gi = gd[h];
    const float gf = gd[H_ + h];
    const float gc = gd[2 * H_ + h];
    const float go = gd[3 * H_ + h];

    const float cn = sigmoidf_(gf) * c[i] + sigmoidf_(gi) * tanhf(gc);
    const float hn = sigmoidf_(go) * tanhf(cn);
    c[i]   = cn;
    h16[i] = (half_t)hn;
    if (dir == 0) hsf[(size_t)t * B_ * H_ + r] = hn;
    else          hsb[(size_t)(S_ - 1 - t) * B_ * H_ + r] = hn;
}

// ---------------------------------------------------------------------------
// Prep kernels
// ---------------------------------------------------------------------------
__global__ __launch_bounds__(256) void embed_tanh(
    const int* __restrict__ seq, const float* __restrict__ table,
    half_t* __restrict__ seq_emb)
{
    const int i  = blockIdx.x * 256 + threadIdx.x;   // < S*B*E
    const int e  = i & (E_ - 1);
    const int sb = i >> 8;                           // E = 256
    const int b  = sb & (B_ - 1);                    // B = 128
    const int s  = sb >> 7;
    const int tok = seq[b * S_ + s];
    const float v = tanhf(table[(size_t)tok * E_ + e]);
    seq_emb[(size_t)s * B_ * E_ + (size_t)b * E_ + e] = (half_t)v;
}

__global__ __launch_bounds__(256) void f32_to_f16_copy(const float* __restrict__ a,
                                                       half_t* __restrict__ o, int n)
{
    const int i = blockIdx.x * 256 + threadIdx.x;
    if (i < n) o[i] = (half_t)a[i];
}

__global__ __launch_bounds__(256) void prep_bias(
    const float* __restrict__ fbi, const float* __restrict__ fbh,
    const float* __restrict__ bbi, const float* __restrict__ bbh,
    float* __restrict__ bias)
{
    const int i = blockIdx.x * 256 + threadIdx.x;
    if (i < FOURH_)           bias[i] = fbi[i] + fbh[i];
    else if (i < 2 * FOURH_) { const int j = i - FOURH_; bias[i] = bbi[j] + bbh[j]; }
}

__global__ __launch_bounds__(256) void zero_state(half_t* __restrict__ h16,
                                                  float* __restrict__ c)
{
    const int i = blockIdx.x * 256 + threadIdx.x;
    if (i < 2 * B_ * H_) { h16[i] = (half_t)0.0f; c[i] = 0.0f; }
}

// Wsum[l,k] = sum_l2 W_cls[l, l2*4H + k]  (the L-fold tiling of feat collapses)
__global__ __launch_bounds__(256) void prep_wsum(const float* __restrict__ Wcls,
                                                 float* __restrict__ wsum)
{
    const int i = blockIdx.x * 256 + threadIdx.x;
    if (i < L_ * FOURH_) {
        const int l = i / FOURH_, k = i - l * FOURH_;
        float s = 0.0f;
        for (int l2 = 0; l2 < L_; ++l2)
            s += Wcls[(size_t)l * L_ * FOURH_ + (size_t)l2 * FOURH_ + k];
        wsum[i] = s;
    }
}

// ---------------------------------------------------------------------------
// pooled[b, 0:2H] = (1/S) * sum_s mask[b,s] * [hs_f | hs_b](s,b,:)
// ---------------------------------------------------------------------------
__global__ __launch_bounds__(256) void pool_mask(
    const int* __restrict__ seq, const float* __restrict__ hsf,
    const float* __restrict__ hsb, float* __restrict__ pooled)
{
    const int i = blockIdx.x * 256 + threadIdx.x;    // < B*2H
    const int b = i >> 10;                            // 2H = 1024
    const int j = i & (2 * H_ - 1);
    float acc = 0.0f;
    for (int t = 0; t < S_; ++t) {
        const float m = (seq[b * S_ + t] > 0) ? 1.0f : 0.0f;
        const float v = (j < H_) ? hsf[(size_t)t * B_ * H_ + (size_t)b * H_ + j]
                                 : hsb[(size_t)t * B_ * H_ + (size_t)b * H_ + (j - H_)];
        acc += m * v;
    }
    pooled[i] = acc * (1.0f / (float)S_);
}

// ---------------------------------------------------------------------------
// logits[b,l] = b_cls[l] + sum_k feat4h[b,k]*Wsum[l,k]; then log_softmax over L.
// feat4h = [pooled(2H) | last(2H)], last = seq_repr[S-1].
// One 256-thread block per batch element.
// ---------------------------------------------------------------------------
__global__ __launch_bounds__(256) void cls_logsoftmax(
    const float* __restrict__ pooled, const float* __restrict__ hsf,
    const float* __restrict__ hsb, const float* __restrict__ wsum,
    const float* __restrict__ bcls, float* __restrict__ out)
{
    const int b = blockIdx.x;
    __shared__ float red[256];
    __shared__ float slog[L_];

    float acc[L_];
#pragma unroll
    for (int l = 0; l < L_; ++l) acc[l] = 0.0f;

    for (int k = threadIdx.x; k < FOURH_; k += 256) {
        float f;
        if (k < 2 * H_) f = pooled[(size_t)b * 2 * H_ + k];
        else {
            const int j = k - 2 * H_;
            f = (j < H_) ? hsf[(size_t)(S_ - 1) * B_ * H_ + (size_t)b * H_ + j]
                         : hsb[(size_t)(S_ - 1) * B_ * H_ + (size_t)b * H_ + (j - H_)];
        }
#pragma unroll
        for (int l = 0; l < L_; ++l) acc[l] += f * wsum[l * FOURH_ + k];
    }

    for (int l = 0; l < L_; ++l) {
        red[threadIdx.x] = acc[l];
        __syncthreads();
        for (int s = 128; s > 0; s >>= 1) {
            if (threadIdx.x < s) red[threadIdx.x] += red[threadIdx.x + s];
            __syncthreads();
        }
        if (threadIdx.x == 0) slog[l] = red[0] + bcls[l];
        __syncthreads();
    }

    if (threadIdx.x == 0) {
        float mx = slog[0];
#pragma unroll
        for (int l = 1; l < L_; ++l) mx = fmaxf(mx, slog[l]);
        float se = 0.0f;
#pragma unroll
        for (int l = 0; l < L_; ++l) se += __expf(slog[l] - mx);
        const float lse = mx + __logf(se);
#pragma unroll
        for (int l = 0; l < L_; ++l) out[(size_t)b * L_ + l] = slog[l] - lse;
    }
}

// Second output: softmax over a singleton axis == exactly 1.0 everywhere.
__global__ __launch_bounds__(256) void fill_ones(float* __restrict__ p, int n)
{
    const int i = blockIdx.x * 256 + threadIdx.x;
    if (i < n) p[i] = 1.0f;
}

// ---------------------------------------------------------------------------
// Host launcher
// ---------------------------------------------------------------------------
extern "C" void kernel_launch(void* const* d_in, const int* in_sizes, int n_in,
                              void* d_out, int out_size, void* d_ws, size_t ws_size,
                              hipStream_t stream)
{
    (void)in_sizes; (void)n_in; (void)out_size; (void)ws_size;

    const int*   seq   = (const int*)  d_in[0];
    const float* embed = (const float*)d_in[1];
    // d_in[2] label_table, d_in[11] attn_p: dead (softmax over singleton axis).
    const float* fW_ih = (const float*)d_in[3];
    const float* fW_hh = (const float*)d_in[4];
    const float* fb_ih = (const float*)d_in[5];
    const float* fb_hh = (const float*)d_in[6];
    const float* bW_ih = (const float*)d_in[7];
    const float* bW_hh = (const float*)d_in[8];
    const float* bb_ih = (const float*)d_in[9];
    const float* bb_hh = (const float*)d_in[10];
    const float* W_cls = (const float*)d_in[12];
    const float* b_cls = (const float*)d_in[13];
    float* out = (float*)d_out;

    char* ws = (char*)d_ws;
    size_t off = 0;
    auto alloc = [&](size_t bytes) {
        size_t o = off;
        off = (off + bytes + 255) & ~(size_t)255;
        return o;
    };

    half_t* seq_emb = (half_t*)(ws + alloc((size_t)S_ * B_ * E_ * sizeof(half_t)));
    half_t* wih16   = (half_t*)(ws + alloc((size_t)2 * FOURH_ * E_ * sizeof(half_t)));
    half_t* whh16   = (half_t*)(ws + alloc((size_t)2 * FOURH_ * H_ * sizeof(half_t)));
    float*  biasws  = (float*) (ws + alloc((size_t)2 * FOURH_ * sizeof(float)));
    half_t* h16     = (half_t*)(ws + alloc((size_t)2 * B_ * H_ * sizeof(half_t)));
    float*  cbuf    = (float*) (ws + alloc((size_t)2 * B_ * H_ * sizeof(float)));
    float*  gbuf    = (float*) (ws + alloc((size_t)2 * B_ * FOURH_ * sizeof(float)));
    float*  hsf     = (float*) (ws + alloc((size_t)S_ * B_ * H_ * sizeof(float)));
    float*  hsb     = (float*) (ws + alloc((size_t)S_ * B_ * H_ * sizeof(float)));
    float*  pooled  = (float*) (ws + alloc((size_t)B_ * 2 * H_ * sizeof(float)));
    float*  wsum    = (float*) (ws + alloc((size_t)L_ * FOURH_ * sizeof(float)));

    // ---- prep: weight/bias conversion, state init, classifier folding ----
    f32_to_f16_copy<<<(FOURH_ * E_) / 256, 256, 0, stream>>>(fW_ih, wih16, FOURH_ * E_);
    f32_to_f16_copy<<<(FOURH_ * E_) / 256, 256, 0, stream>>>(bW_ih, wih16 + (size_t)FOURH_ * E_, FOURH_ * E_);
    f32_to_f16_copy<<<(FOURH_ * H_) / 256, 256, 0, stream>>>(fW_hh, whh16, FOURH_ * H_);
    f32_to_f16_copy<<<(FOURH_ * H_) / 256, 256, 0, stream>>>(bW_hh, whh16 + (size_t)FOURH_ * H_, FOURH_ * H_);
    prep_bias<<<(2 * FOURH_) / 256, 256, 0, stream>>>(fb_ih, fb_hh, bb_ih, bb_hh, biasws);
    zero_state<<<(2 * B_ * H_) / 256, 256, 0, stream>>>(h16, cbuf);
    prep_wsum<<<(L_ * FOURH_ + 255) / 256, 256, 0, stream>>>(W_cls, wsum);

    // ---- embedding + tanh into (S,B,E) f16 ----
    embed_tanh<<<((size_t)S_ * B_ * E_) / 256, 256, 0, stream>>>(seq, embed, seq_emb);

    // ---- bidirectional LSTM scan: 256 steps, both directions fused ----
    for (int t = 0; t < S_; ++t) {
        lstm_step_gemm<<<128, 256, 0, stream>>>(seq_emb, wih16, whh16, h16, biasws, gbuf, t);
        lstm_gates<<<(2 * B_ * H_) / 256, 256, 0, stream>>>(gbuf, cbuf, h16, hsf, hsb, t);
    }

    // ---- masked mean pooling over s ----
    pool_mask<<<(B_ * 2 * H_) / 256, 256, 0, stream>>>(seq, hsf, hsb, pooled);

    // ---- classifier + log_softmax -> out[0 : B*L] ----
    cls_logsoftmax<<<B_, 256, 0, stream>>>(pooled, hsf, hsb, wsum, b_cls, out);

    // ---- attention weights output: exactly ones -> out[B*L : B*L + B*L*S] ----
    fill_ones<<<(B_ * L_ * S_ + 255) / 256, 256, 0, stream>>>(out + B_ * L_, B_ * L_ * S_);
}
